// MultiHeadAttention_81209241633560
// MI455X (gfx1250) — matrix operand order repaired and checked by reading, not compiled, not run on hardware
//
#include <hip/hip_runtime.h>
#include <hip/hip_bf16.h>

// ---------------------------------------------------------------------------
// MI455X (gfx1250) fused MHA: f16 convert pass -> KQV-GEMM+RoPE -> flash
// attention -> projection. All matrix math via v_wmma_f32_16x16x32_f16
// (f16 operands, fp32 accumulate). Wave32. GEMM waves own 32x32 output
// blocks (2x2 WMMA tiles) so each pair of A/B fragments feeds 4 WMMAs.
// ---------------------------------------------------------------------------

typedef __attribute__((ext_vector_type(16))) _Float16 v16h;
typedef __attribute__((ext_vector_type(8)))  _Float16 v8h;
typedef __attribute__((ext_vector_type(8)))  float    v8f;
typedef __attribute__((ext_vector_type(4)))  float    f32x4;

#define NUM_HEADS 16
#define HEAD_SIZE 64
#define SEQ_T     2048
#define DIM_D     1024
#define BATCH_B   4

// ---- Fragment loaders (f16 sources only; no conversions in hot loops) ----
// A-fragment (16x32, M x K): lane holds row (lane&15); K = kb..kb+7 and
// kb+16..kb+23 where kb = 8*(lane>=16).  (ISA 7.12.2, 16-bit A table.)
__device__ __forceinline__ v16h afrag_from_h(const _Float16* __restrict__ rowk, int half) {
  const _Float16* p = rowk + half * 8;
  v8h c0 = *(const v8h*)(p);
  v8h c1 = *(const v8h*)(p + 16);
  v16h a;
#pragma unroll
  for (int j = 0; j < 8; ++j) { a[j] = c0[j]; a[j + 8] = c1[j]; }
  return a;
}
// B-fragment (32x16, K x N): lane holds column (lane&15); K = kb..kb+7 and
// kb+16..kb+23 where kb = 8*(lane>=16) (same K order as the A-fragment).
__device__ __forceinline__ v16h bfrag_from_h(const _Float16* __restrict__ colk, int half) {
  const _Float16* p = colk + half * 8;
  v8h c0 = *(const v8h*)(p);
  v8h c1 = *(const v8h*)(p + 16);
  v16h b;
#pragma unroll
  for (int j = 0; j < 8; ++j) { b[j] = c0[j]; b[j + 8] = c1[j]; }
  return b;
}

#define WMMA_F16(A, B, C) \
  __builtin_amdgcn_wmma_f32_16x16x32_f16(false, (A), false, (B), (short)0, (C), false, false)

// ---------------------------------------------------------------------------
// Kernel 0: bulk fp32 -> f16 conversion (pure bandwidth, ~3us total at
// 23.3 TB/s for x + both weights). 8 elements per thread, b128 traffic.
// ---------------------------------------------------------------------------
__global__ void __launch_bounds__(256) cvt_f32_to_f16_kernel(
    const float* __restrict__ src, _Float16* __restrict__ dst)
{
  size_t i = ((size_t)blockIdx.x * 256 + threadIdx.x) * 8;
  f32x4 a = *(const f32x4*)(src + i);
  f32x4 b = *(const f32x4*)(src + i + 4);
  v8h o;
#pragma unroll
  for (int j = 0; j < 4; ++j) { o[j] = (_Float16)a[j]; o[4 + j] = (_Float16)b[j]; }
  *(v8h*)(dst + i) = o;
}

// ---------------------------------------------------------------------------
// Kernel 1: kqv = x @ w_kqv^T (8192x1024 * 1024x3072), f16 inputs, fused
// RoPE on k,q; scatter to head-major f16 layouts (v transposed [B,H,hs,T]).
// Each wave: 32x32 output block = 2x2 WMMA tiles.
// ---------------------------------------------------------------------------
__device__ __forceinline__ void kqv_epilogue(
    v8f c, int mbase, int nbase, int lrow, int half,
    _Float16* __restrict__ qh, _Float16* __restrict__ kh, _Float16* __restrict__ vT)
{
  const int sec    = nbase >> 10;     // [k | q | v]
  const int within = nbase & 1023;
  const int h      = within >> 6;
  const int dbase  = within & 63;
  const int d      = dbase + lrow;

  if (sec == 2) {                     // v -> [B,H,hs,T]
#pragma unroll
    for (int r = 0; r < 8; ++r) {
      int mg = mbase + r + 8 * half;
      int bb = mg >> 11, t = mg & (SEQ_T - 1);
      vT[(((size_t)bb * NUM_HEADS + h) * HEAD_SIZE + d) * SEQ_T + t] = (_Float16)c[r];
    }
  } else {                            // k (0) / q (1) with RoPE
    const int   i    = d >> 1;
    const float freq = __expf((float)i * -0.28782313662425572f); // 10000^(-2i/64)
    const float sgn  = (d & 1) ? 1.0f : -1.0f;
    _Float16* dst = (sec == 1) ? qh : kh;
#pragma unroll
    for (int r = 0; r < 8; ++r) {
      int mg = mbase + r + 8 * half;
      int bb = mg >> 11, t = mg & (SEQ_T - 1);
      float sn, cs;
      __sincosf((float)t * freq, &sn, &cs);
      float other = __shfl_xor(c[r], 1, 32);  // even/odd column partner
      float val   = c[r] * cs + sgn * other * sn;
      dst[(((size_t)bb * NUM_HEADS + h) * SEQ_T + t) * HEAD_SIZE + d] = (_Float16)val;
    }
  }
}

__global__ void __launch_bounds__(256) kqv_rope_kernel(
    const _Float16* __restrict__ xh, const _Float16* __restrict__ wh,
    _Float16* __restrict__ qh, _Float16* __restrict__ kh, _Float16* __restrict__ vT)
{
  const int lane = threadIdx.x & 31, lrow = lane & 15, half = lane >> 4;
  const int gw = blockIdx.x * 8 + (threadIdx.x >> 5);
  const int tn = gw % 96, tm = gw / 96;       // 256 x 96 blocks of 32x32
  const int mbase = tm * 32, nbase = tn * 32;

  const _Float16* a0row = xh + (size_t)(mbase +      lrow) * DIM_D;
  const _Float16* a1row = xh + (size_t)(mbase + 16 + lrow) * DIM_D;
  const _Float16* b0row = wh + (size_t)(nbase +      lrow) * DIM_D;
  const _Float16* b1row = wh + (size_t)(nbase + 16 + lrow) * DIM_D;

  v8f c00 = {}, c01 = {}, c10 = {}, c11 = {};
#pragma unroll 2
  for (int k = 0; k < DIM_D; k += 32) {
    v16h a0 = afrag_from_h(a0row + k, half);
    v16h a1 = afrag_from_h(a1row + k, half);
    v16h b0 = bfrag_from_h(b0row + k, half);
    v16h b1 = bfrag_from_h(b1row + k, half);
    c00 = WMMA_F16(a0, b0, c00);
    c01 = WMMA_F16(a0, b1, c01);
    c10 = WMMA_F16(a1, b0, c10);
    c11 = WMMA_F16(a1, b1, c11);
  }
  kqv_epilogue(c00, mbase,      nbase,      lrow, half, qh, kh, vT);
  kqv_epilogue(c01, mbase,      nbase + 16, lrow, half, qh, kh, vT);
  kqv_epilogue(c10, mbase + 16, nbase,      lrow, half, qh, kh, vT);
  kqv_epilogue(c11, mbase + 16, nbase + 16, lrow, half, qh, kh, vT);
}

// ---------------------------------------------------------------------------
// Kernel 2: flash attention. Block = 8 waves; wave = 16 q-rows of one (b,h).
// Online softmax over 32-key tiles; P re-laid-out via per-wave LDS slab.
// q/k/vT (48 MB f16) fit in the 192 MB L2. Output written as f16 so the
// projection GEMM loads fragments without conversion.
// ---------------------------------------------------------------------------
__global__ void __launch_bounds__(256) attn_kernel(
    const _Float16* __restrict__ qh, const _Float16* __restrict__ kh,
    const _Float16* __restrict__ vT, _Float16* __restrict__ attnh)
{
  __shared__ _Float16 plds[8][16][32];          // per-wave P staging (1 KB each)
  const int lane = threadIdx.x & 31, lrow = lane & 15, half = lane >> 4;
  const int wid  = threadIdx.x >> 5;
  const int bh   = blockIdx.y;                  // b*16 + h
  const int bb   = bh >> 4, h = bh & 15;
  const int qbase = blockIdx.x * 128 + wid * 16;

  const _Float16* qhead = qh + (size_t)bh * SEQ_T * HEAD_SIZE;
  const _Float16* khead = kh + (size_t)bh * SEQ_T * HEAD_SIZE;
  const _Float16* vhead = vT + (size_t)bh * HEAD_SIZE * SEQ_T;

  const v16h aq0 = afrag_from_h(qhead + (size_t)(qbase + lrow) * HEAD_SIZE + 0,  half);
  const v16h aq1 = afrag_from_h(qhead + (size_t)(qbase + lrow) * HEAD_SIZE + 32, half);

  v8f o0 = {}, o1 = {}, o2 = {}, o3 = {};
  float mrow[8], lsum[8];
#pragma unroll
  for (int r = 0; r < 8; ++r) { mrow[r] = -1e30f; lsum[r] = 0.0f; }

  const int ktiles = (qbase + 16 + 31) >> 5;
  for (int jt = 0; jt < ktiles; ++jt) {
    const int jbase = jt * 32;

    // S = Q K^T : two 16-key column subtiles, K(=hs) split 2x32
    v16h bk00 = bfrag_from_h(khead + (size_t)(jbase +      lrow) * HEAD_SIZE + 0,  half);
    v16h bk01 = bfrag_from_h(khead + (size_t)(jbase +      lrow) * HEAD_SIZE + 32, half);
    v16h bk10 = bfrag_from_h(khead + (size_t)(jbase + 16 + lrow) * HEAD_SIZE + 0,  half);
    v16h bk11 = bfrag_from_h(khead + (size_t)(jbase + 16 + lrow) * HEAD_SIZE + 32, half);
    v8f s0 = {}, s1 = {};
    s0 = WMMA_F16(aq0, bk00, s0);
    s0 = WMMA_F16(aq1, bk01, s0);
    s1 = WMMA_F16(aq0, bk10, s1);
    s1 = WMMA_F16(aq1, bk11, s1);

    const int j0 = jbase + lrow, j1 = jbase + 16 + lrow;
    float p0[8], p1[8], alpha[8];
#pragma unroll
    for (int r = 0; r < 8; ++r) {
      int mg = qbase + r + 8 * half;
      float f0 = (j0 <= mg) ? s0[r] * 0.125f : -1e30f;   // 1/sqrt(64)
      float f1 = (j1 <= mg) ? s1[r] * 0.125f : -1e30f;
      float mx = fmaxf(f0, f1);                           // row lives on 16 lanes
      mx = fmaxf(mx, __shfl_xor(mx, 1, 32));
      mx = fmaxf(mx, __shfl_xor(mx, 2, 32));
      mx = fmaxf(mx, __shfl_xor(mx, 4, 32));
      mx = fmaxf(mx, __shfl_xor(mx, 8, 32));
      float mn = fmaxf(mrow[r], mx);
      float al = __expf(mrow[r] - mn);
      float e0 = __expf(f0 - mn);
      float e1 = __expf(f1 - mn);
      float rs = e0 + e1;
      rs += __shfl_xor(rs, 1, 32);
      rs += __shfl_xor(rs, 2, 32);
      rs += __shfl_xor(rs, 4, 32);
      rs += __shfl_xor(rs, 8, 32);
      lsum[r] = lsum[r] * al + rs;
      mrow[r] = mn;
      alpha[r] = al;
      p0[r] = e0; p1[r] = e1;
    }
#pragma unroll
    for (int r = 0; r < 8; ++r) {
      o0[r] *= alpha[r]; o1[r] *= alpha[r]; o2[r] *= alpha[r]; o3[r] *= alpha[r];
    }

    // C-layout P -> LDS [row m][key kk], read back as A-fragment (wave-local;
    // the compiler orders the ds_store/ds_load pair with s_wait_dscnt).
#pragma unroll
    for (int r = 0; r < 8; ++r) {
      int m = r + 8 * half;
      plds[wid][m][lrow]      = (_Float16)p0[r];
      plds[wid][m][16 + lrow] = (_Float16)p1[r];
    }
    v16h ap = afrag_from_h(&plds[wid][lrow][0], half);

    // O += P @ V : vT rows are d-major so each lane's K-values are two contiguous runs of 8
    v16h bv0 = bfrag_from_h(vhead + (size_t)( 0 + lrow) * SEQ_T + jbase, half);
    v16h bv1 = bfrag_from_h(vhead + (size_t)(16 + lrow) * SEQ_T + jbase, half);
    v16h bv2 = bfrag_from_h(vhead + (size_t)(32 + lrow) * SEQ_T + jbase, half);
    v16h bv3 = bfrag_from_h(vhead + (size_t)(48 + lrow) * SEQ_T + jbase, half);
    o0 = WMMA_F16(ap, bv0, o0);
    o1 = WMMA_F16(ap, bv1, o1);
    o2 = WMMA_F16(ap, bv2, o2);
    o3 = WMMA_F16(ap, bv3, o3);
  }

  // Epilogue: normalize and scatter to [B,T,H*hs] as f16
  _Float16* outh = attnh + (size_t)bb * SEQ_T * DIM_D + (size_t)h * HEAD_SIZE;
#pragma unroll
  for (int r = 0; r < 8; ++r) {
    int mg = qbase + r + 8 * half;
    float inv = 1.0f / lsum[r];
    _Float16* row = outh + (size_t)mg * DIM_D;
    row[ 0 + lrow] = (_Float16)(o0[r] * inv);
    row[16 + lrow] = (_Float16)(o1[r] * inv);
    row[32 + lrow] = (_Float16)(o2[r] * inv);
    row[48 + lrow] = (_Float16)(o3[r] * inv);
  }
}

// ---------------------------------------------------------------------------
// Kernel 3: out = attn @ w_proj^T + b_proj (8192x1024 * 1024x1024), f16
// inputs, fp32 output. 32x32 block per wave.
// ---------------------------------------------------------------------------
__global__ void __launch_bounds__(256) proj_kernel(
    const _Float16* __restrict__ attnh, const _Float16* __restrict__ wh,
    const float* __restrict__ bias, float* __restrict__ out)
{
  const int lane = threadIdx.x & 31, lrow = lane & 15, half = lane >> 4;
  const int gw = blockIdx.x * 8 + (threadIdx.x >> 5);
  const int tn = gw & 31, tm = gw >> 5;       // 256 x 32 blocks of 32x32
  const int mbase = tm * 32, nbase = tn * 32;

  const _Float16* a0row = attnh + (size_t)(mbase +      lrow) * DIM_D;
  const _Float16* a1row = attnh + (size_t)(mbase + 16 + lrow) * DIM_D;
  const _Float16* b0row = wh    + (size_t)(nbase +      lrow) * DIM_D;
  const _Float16* b1row = wh    + (size_t)(nbase + 16 + lrow) * DIM_D;

  v8f c00 = {}, c01 = {}, c10 = {}, c11 = {};
#pragma unroll 2
  for (int k = 0; k < DIM_D; k += 32) {
    v16h a0 = afrag_from_h(a0row + k, half);
    v16h a1 = afrag_from_h(a1row + k, half);
    v16h b0 = bfrag_from_h(b0row + k, half);
    v16h b1 = bfrag_from_h(b1row + k, half);
    c00 = WMMA_F16(a0, b0, c00);
    c01 = WMMA_F16(a0, b1, c01);
    c10 = WMMA_F16(a1, b0, c10);
    c11 = WMMA_F16(a1, b1, c11);
  }
  float bv0 = bias[nbase + lrow];
  float bv1 = bias[nbase + 16 + lrow];
#pragma unroll
  for (int r = 0; r < 8; ++r) {
    int mg0 = mbase + r + 8 * half;
    int mg1 = mg0 + 16;
    out[(size_t)mg0 * DIM_D + nbase +      lrow] = c00[r] + bv0;
    out[(size_t)mg0 * DIM_D + nbase + 16 + lrow] = c01[r] + bv1;
    out[(size_t)mg1 * DIM_D + nbase +      lrow] = c10[r] + bv0;
    out[(size_t)mg1 * DIM_D + nbase + 16 + lrow] = c11[r] + bv1;
  }
}

// ---------------------------------------------------------------------------
extern "C" void kernel_launch(void* const* d_in, const int* in_sizes, int n_in,
                              void* d_out, int out_size, void* d_ws, size_t ws_size,
                              hipStream_t stream) {
  (void)in_sizes; (void)n_in; (void)out_size; (void)ws_size;
  const float* x      = (const float*)d_in[0];
  const float* w_kqv  = (const float*)d_in[1];
  const float* w_proj = (const float*)d_in[2];
  const float* b_proj = (const float*)d_in[3];
  float* out = (float*)d_out;

  char* ws = (char*)d_ws;
  const size_t NX  = (size_t)BATCH_B * SEQ_T * DIM_D;   // 8 M elements
  const size_t NWK = (size_t)3 * DIM_D * DIM_D;         // 3 M
  const size_t NWP = (size_t)DIM_D * DIM_D;             // 1 M
  const size_t HSZ = (size_t)BATCH_B * NUM_HEADS * SEQ_T * HEAD_SIZE * sizeof(_Float16); // 16 MB

  size_t off = 0;
  _Float16* xh    = (_Float16*)(ws + off); off += NX  * sizeof(_Float16);  // 16 MB
  _Float16* wkh   = (_Float16*)(ws + off); off += NWK * sizeof(_Float16);  //  6 MB
  _Float16* wph   = (_Float16*)(ws + off); off += NWP * sizeof(_Float16);  //  2 MB
  _Float16* qh    = (_Float16*)(ws + off); off += HSZ;                     // 16 MB
  _Float16* kh    = (_Float16*)(ws + off); off += HSZ;                     // 16 MB
  _Float16* vT    = (_Float16*)(ws + off); off += HSZ;                     // 16 MB
  _Float16* attnh = (_Float16*)(ws + off);                                 // 16 MB

  // One-shot f32->f16 conversion (amortized across all tile re-reads)
  cvt_f32_to_f16_kernel<<<dim3(NX  / (256 * 8)), 256, 0, stream>>>(x, xh);
  cvt_f32_to_f16_kernel<<<dim3(NWK / (256 * 8)), 256, 0, stream>>>(w_kqv, wkh);
  cvt_f32_to_f16_kernel<<<dim3(NWP / (256 * 8)), 256, 0, stream>>>(w_proj, wph);

  // 256*96 blocks of 32x32, 8 waves/block
  kqv_rope_kernel<<<dim3((256 * 96) / 8), 256, 0, stream>>>(xh, wkh, qh, kh, vT);
  // 16 q-blocks of 128 rows per (b,h); 64 (b,h) pairs
  attn_kernel<<<dim3(16, 64), 256, 0, stream>>>(qh, kh, vT, attnh);
  // 256*32 blocks of 32x32, 8 waves/block
  proj_kernel<<<dim3((256 * 32) / 8), 256, 0, stream>>>(attnh, wph, b_proj, out);
}
